// PointNetPP_69449621176721
// MI455X (gfx1250) — matrix operand order repaired and checked by reading, not compiled, not run on hardware
//
#include <hip/hip_runtime.h>

typedef _Float16 f16;
typedef __attribute__((ext_vector_type(16))) _Float16 v16h;
typedef __attribute__((ext_vector_type(8)))  float    v8f;

#define NBATCH 2
#define GPAD 16  // padded column count of grouped f16 input (cin <= 15)

// ---------------------------------------------------------------- WMMA core
__device__ __forceinline__ v8f wmma16(v16h a, v16h b, v8f c) {
  // D = A(16x32 f16) * B(32x16 f16) + C(16x16 f32)
  return __builtin_amdgcn_wmma_f32_16x16x32_f16(false, a, false, b, (short)0, c,
                                                false, false);
}

// order-preserving float<->uint mapping for atomicMax on signed floats
__device__ __forceinline__ unsigned fkey(float f) {
  unsigned b = __float_as_uint(f);
  return b ^ ((unsigned)((int)b >> 31) | 0x80000000u);
}
__device__ __forceinline__ float finv(unsigned u) {
  unsigned b = (u & 0x80000000u) ? (u ^ 0x80000000u) : ~u;
  return __uint_as_float(b);
}

// A fragment (16x32 f16) from row-major f16 buffer, leading dim ld.
// ISA layout: lane half hl, element i -> K = (i<8 ? hl*8+i : 16+hl*8+(i-8))
__device__ __forceinline__ v16h load_a_f16(const f16* base, int row0, int k0,
                                           int ld, int lane) {
  const int lr = lane & 15, hl = lane >> 4;
  const f16* p = base + (size_t)(row0 + lr) * ld;
  v16h a;
#pragma unroll
  for (int i = 0; i < 16; i += 2) {
    int K = k0 + ((i < 8) ? (hl * 8 + i) : (16 + hl * 8 + (i - 8)));
    a[i]     = p[K];
    a[i + 1] = p[K + 1];
  }
  return a;
}

// A fragment from padded-16-column f16 rows (K tile 32; K>=16 is always zero).
// i<8 -> K = hl*8+i in [0,16): one contiguous 16B load per lane. i>=8 -> 0.
__device__ __forceinline__ v16h load_a_g16(const f16* base, int row0, int lane) {
  const int lr = lane & 15, hl = lane >> 4;
  const f16* p = base + (size_t)(row0 + lr) * GPAD + hl * 8;
  v16h a;
#pragma unroll
  for (int i = 0; i < 8; i += 2) { a[i] = p[i]; a[i + 1] = p[i + 1]; }
#pragma unroll
  for (int i = 8; i < 16; ++i) a[i] = (f16)0.0f;
  return a;
}

// B fragment (32x16 f16) from transposed weights Wt[N][ldk] (row n = output col)
// ISA layout: lane lr = N, element i -> K = (i<8 ? hl*8+i : 16+hl*8+(i-8))
__device__ __forceinline__ v16h load_b_t(const f16* wt, int n0, int k0, int ldk,
                                         int lane) {
  const int lr = lane & 15, hl = lane >> 4;
  const f16* p = wt + (size_t)(n0 + lr) * ldk + k0;
  v16h b;
#pragma unroll
  for (int i = 0; i < 16; ++i) b[i] = p[(i < 8) ? (hl * 8 + i) : (16 + hl * 8 + (i - 8))];
  return b;
}

// ---------------------------------------------------------------- PointNet
// One block (4 wave32) per group. gin: (G, m, GPAD) f16. feats: (G, COUT) f32.
// Weights pre-transposed f16: w1t[64][32], wt1t[128][64], wt2t[4096][128],
// w2t[128][64], w3t[ceil16(COUT)][128].
template <int COUT>
__global__ void __launch_bounds__(128)
pointnet_kernel(const f16* __restrict__ gin,
                const float* __restrict__ b1, const float* __restrict__ bt1,
                const float* __restrict__ bt2, const float* __restrict__ b2,
                const float* __restrict__ b3,
                const f16* __restrict__ w1t,  const f16* __restrict__ wt1t,
                const f16* __restrict__ wt2t, const f16* __restrict__ w2t,
                const f16* __restrict__ w3t,
                float* __restrict__ feats, int m) {
  extern __shared__ char smem_raw[];
  f16* hbuf = (f16*)smem_raw;              // m x 64 (h, later overwritten by h2)
  f16* Tt   = hbuf + (size_t)m * 64;       // 64 x 64 transposed: Tt[d][c]=T[c][d]
  f16* sbuf = Tt + 64 * 64;                // 4 waves x (16 x 128) staging
  unsigned* tmax = (unsigned*)(sbuf + 4 * 16 * 128);  // 128
  float*    tvec = (float*)(tmax + 128);              // 128
  unsigned* fmax = (unsigned*)(tvec + 128);           // 256

  const int tid  = threadIdx.x;
  const int lane = tid & 31;
  const int wv   = tid >> 5;
  const int lr   = lane & 15, hl = lane >> 4;
  const int g    = blockIdx.x;
  const int nrt  = m >> 4;
  const f16* gr  = gin + (size_t)g * m * GPAD;
  const v8f vzero = {0.f, 0.f, 0.f, 0.f, 0.f, 0.f, 0.f, 0.f};

  if (tid < 128) tmax[tid] = 0u;
  for (int i = tid; i < 256; i += 128) fmax[i] = 0x007FFFFFu;  // fkey(-inf)
  __syncthreads();

  // ---- A+B per row tile: h = relu(x@w1+b1);  t = colmax(relu(h@wt1+bt1))
  for (int rt = wv; rt < nrt; rt += 4) {
    v16h ax = load_a_g16(gr, rt * 16, lane);
#pragma unroll
    for (int nt = 0; nt < 4; ++nt) {
      v8f acc = vzero;
      acc = wmma16(ax, load_b_t(w1t, nt * 16, 0, 32, lane), acc);
      const int col = nt * 16 + lr;
      const float bias = b1[col];
#pragma unroll
      for (int r = 0; r < 8; ++r)
        hbuf[(rt * 16 + r + 8 * hl) * 64 + col] = (f16)fmaxf(acc[r] + bias, 0.f);
    }
#pragma unroll
    for (int nt = 0; nt < 8; ++nt) {
      v8f acc = vzero;
#pragma unroll
      for (int ks = 0; ks < 64; ks += 32) {
        v16h aa = load_a_f16(hbuf, rt * 16, ks, 64, lane);
        acc = wmma16(aa, load_b_t(wt1t, nt * 16, ks, 64, lane), acc);
      }
      const int col = nt * 16 + lr;
      const float bias = bt1[col];
      float mx = 0.f;  // relu floor
#pragma unroll
      for (int r = 0; r < 8; ++r) mx = fmaxf(mx, acc[r] + bias);
      mx = fmaxf(mx, __shfl_xor(mx, 16, 32));
      if (hl == 0) atomicMax(&tmax[col], __float_as_uint(mx));  // >=0: raw bits ok
    }
  }
  __syncthreads();
  if (tid < 128) tvec[tid] = __uint_as_float(tmax[tid]);
  __syncthreads();

  // ---- C: T = (t @ wt2 + bt2).reshape(64,64) + I, stored transposed (VALU matvec)
  for (int o = tid; o < 4096; o += 128) {
    const f16* wr = wt2t + (size_t)o * 128;
    float s = bt2[o];
    for (int kk = 0; kk < 128; ++kk) s += tvec[kk] * (float)wr[kk];
    int r = o >> 6, c = o & 63;
    if (r == c) s += 1.0f;
    Tt[c * 64 + r] = (f16)s;
  }
  __syncthreads();

  // ---- D/E/F per row tile
  constexpr int NCT = (COUT + 15) >> 4;
  f16* sb = sbuf + wv * (16 * 128);
  for (int rt = wv; rt < nrt; rt += 4) {
    // D: h2 = h @ T  (row-local overwrite of hbuf)
    v16h a0 = load_a_f16(hbuf, rt * 16, 0, 64, lane);
    v16h a1 = load_a_f16(hbuf, rt * 16, 32, 64, lane);
    v8f accs[4];
#pragma unroll
    for (int nt = 0; nt < 4; ++nt) {
      v8f acc = vzero;
      acc = wmma16(a0, load_b_t(Tt, nt * 16, 0, 64, lane), acc);
      acc = wmma16(a1, load_b_t(Tt, nt * 16, 32, 64, lane), acc);
      accs[nt] = acc;
    }
#pragma unroll
    for (int nt = 0; nt < 4; ++nt) {
      const int col = nt * 16 + lr;
#pragma unroll
      for (int r = 0; r < 8; ++r)
        hbuf[(rt * 16 + r + 8 * hl) * 64 + col] = (f16)accs[nt][r];
    }
    // E: h3 = relu(h2 @ w2 + b2) -> per-wave staging tile
    v16h c0 = load_a_f16(hbuf, rt * 16, 0, 64, lane);
    v16h c1 = load_a_f16(hbuf, rt * 16, 32, 64, lane);
#pragma unroll
    for (int nt = 0; nt < 8; ++nt) {
      v8f acc = vzero;
      acc = wmma16(c0, load_b_t(w2t, nt * 16, 0, 64, lane), acc);
      acc = wmma16(c1, load_b_t(w2t, nt * 16, 32, 64, lane), acc);
      const int col = nt * 16 + lr;
      const float bias = b2[col];
#pragma unroll
      for (int r = 0; r < 8; ++r)
        sb[(r + 8 * hl) * 128 + col] = (f16)fmaxf(acc[r] + bias, 0.f);
    }
    // F: o = h3 @ w3 + b3; running column max over rows
    v16h f0 = load_a_f16(sb, 0, 0, 128, lane);
    v16h f1 = load_a_f16(sb, 0, 32, 128, lane);
    v16h f2 = load_a_f16(sb, 0, 64, 128, lane);
    v16h f3 = load_a_f16(sb, 0, 96, 128, lane);
#pragma unroll
    for (int ct = 0; ct < NCT; ++ct) {
      v8f acc = vzero;
      acc = wmma16(f0, load_b_t(w3t, ct * 16, 0, 128, lane), acc);
      acc = wmma16(f1, load_b_t(w3t, ct * 16, 32, 128, lane), acc);
      acc = wmma16(f2, load_b_t(w3t, ct * 16, 64, 128, lane), acc);
      acc = wmma16(f3, load_b_t(w3t, ct * 16, 96, 128, lane), acc);
      const int col = ct * 16 + lr;
      const float bias = (col < COUT) ? b3[col] : 0.f;
      float mx = acc[0] + bias;
#pragma unroll
      for (int r = 1; r < 8; ++r) mx = fmaxf(mx, acc[r] + bias);
      mx = fmaxf(mx, __shfl_xor(mx, 16, 32));
      if (hl == 0 && col < COUT) atomicMax(&fmax[col], fkey(mx));
    }
  }
  __syncthreads();
  for (int i = tid; i < COUT; i += 128)
    feats[(size_t)g * COUT + i] = finv(fmax[i]);
}

// ---------------------------------------------------------------- helpers
__global__ void zero_u32(unsigned* p, int n) {
  int i = blockIdx.x * blockDim.x + threadIdx.x;
  if (i < n) p[i] = 0u;
}

// w[K][N] (f32 row-major) -> wt[Npad][Kpad] f16, zero-padded
__global__ void conv_w_t(const float* __restrict__ w, f16* __restrict__ wt,
                         int K, int N, int Kpad, int Npad) {
  int idx = blockIdx.x * blockDim.x + threadIdx.x;
  if (idx >= Npad * Kpad) return;
  int n = idx / Kpad, kk = idx % Kpad;
  float v = (n < N && kk < K) ? w[(size_t)kk * N + n] : 0.f;
  wt[idx] = (f16)v;
}

// Farthest point sampling (matches jnp.argmax first-index tie-break). 1 block/batch.
__global__ void fps_kernel(const float* __restrict__ x, int m, int stride, int k,
                           float* __restrict__ cents) {
  const int b = blockIdx.x, t = threadIdx.x;
  __shared__ float d[512];
  __shared__ float sval[256];
  __shared__ int   sidx[256];
  __shared__ float c0, c1, c2;
  __shared__ int   bi;
  const float* xb = x + (size_t)b * m * stride;
  const float p0x = xb[0], p0y = xb[1], p0z = xb[2];
  for (int j = t; j < m; j += 256) {
    float dx = xb[j * stride] - p0x, dy = xb[j * stride + 1] - p0y,
          dz = xb[j * stride + 2] - p0z;
    d[j] = sqrtf(dx * dx + dy * dy + dz * dz);
  }
  if (t == 0) {
    d[0] = -__builtin_inff();
    cents[(size_t)b * k * 3 + 0] = p0x;
    cents[(size_t)b * k * 3 + 1] = p0y;
    cents[(size_t)b * k * 3 + 2] = p0z;
  }
  __syncthreads();
  for (int it = 1; it < k; ++it) {
    float bv = -__builtin_inff();
    int bj = 0x7fffffff;
    for (int j = t; j < m; j += 256) {
      float v = d[j];
      if (v > bv) { bv = v; bj = j; }
    }
    sval[t] = bv; sidx[t] = bj;
    __syncthreads();
    for (int s = 128; s > 0; s >>= 1) {
      if (t < s) {
        float ov = sval[t + s]; int oj = sidx[t + s];
        if (ov > sval[t] || (ov == sval[t] && oj < sidx[t])) {
          sval[t] = ov; sidx[t] = oj;
        }
      }
      __syncthreads();
    }
    if (t == 0) {
      bi = sidx[0];
      c0 = xb[bi * stride]; c1 = xb[bi * stride + 1]; c2 = xb[bi * stride + 2];
      cents[((size_t)b * k + it) * 3 + 0] = c0;
      cents[((size_t)b * k + it) * 3 + 1] = c1;
      cents[((size_t)b * k + it) * 3 + 2] = c2;
    }
    __syncthreads();
    for (int j = t; j < m; j += 256) {
      if (j == bi) d[j] = -__builtin_inff();
      else {
        float dx = xb[j * stride] - c0, dy = xb[j * stride + 1] - c1,
              dz = xb[j * stride + 2] - c2;
        d[j] = fminf(d[j], sqrtf(dx * dx + dy * dy + dz * dz));
      }
    }
    __syncthreads();
  }
}

__global__ void assign_kernel(const float* __restrict__ x,
                              const float* __restrict__ cents,
                              int* __restrict__ grp, int* __restrict__ rnk,
                              int* __restrict__ cnts, int m, int stride, int k) {
  int idx = blockIdx.x * blockDim.x + threadIdx.x;
  if (idx >= NBATCH * m) return;
  int b = idx / m, j = idx % m;
  const float* p = x + ((size_t)b * m + j) * stride;
  const float px = p[0], py = p[1], pz = p[2];
  const float* cb = cents + (size_t)b * k * 3;
  float best = __builtin_inff(); int bg = 0;
  for (int i = 0; i < k; ++i) {
    float dx = px - cb[i * 3], dy = py - cb[i * 3 + 1], dz = pz - cb[i * 3 + 2];
    float d2 = dx * dx + dy * dy + dz * dz;
    if (d2 < best) { best = d2; bg = i; }
  }
  grp[idx] = bg;
  rnk[idx] = atomicAdd(&cnts[b * k + bg], 1);
}

// grouped[g][slot] = (-cent, 0...) base, f16 padded to GPAD columns
__global__ void init_grouped(f16* __restrict__ gout,
                             const float* __restrict__ cents, int k, int m) {
  int idx = blockIdx.x * blockDim.x + threadIdx.x;
  if (idx >= NBATCH * k * m) return;
  int g = idx / m, slot = idx % m;
  f16* row = gout + ((size_t)g * m + slot) * GPAD;
  const float* ce = cents + (size_t)g * 3;
  row[0] = (f16)(-ce[0]); row[1] = (f16)(-ce[1]); row[2] = (f16)(-ce[2]);
#pragma unroll
  for (int i = 3; i < GPAD; ++i) row[i] = (f16)0.0f;
}

__global__ void scatter_kernel(const float* __restrict__ x,
                               const int* __restrict__ grp,
                               const int* __restrict__ rnk,
                               f16* __restrict__ gout, int m, int c, int k) {
  int idx = blockIdx.x * blockDim.x + threadIdx.x;
  if (idx >= NBATCH * m) return;
  int b = idx / m, j = idx % m;
  int g = b * k + grp[idx];
  const float* src = x + ((size_t)b * m + j) * c;
  f16* dst = gout + ((size_t)g * m + rnk[idx]) * GPAD;
  for (int i = 0; i < c; ++i) dst[i] = (f16)src[i];
  for (int i = c; i < GPAD; ++i) dst[i] = (f16)0.0f;
}

// pack f32 rows (rows x c) into padded f16 rows (rows x GPAD)
__global__ void pack_rows_f16(const float* __restrict__ x, f16* __restrict__ out,
                              int rows, int c) {
  int idx = blockIdx.x * blockDim.x + threadIdx.x;
  if (idx >= rows) return;
  const float* s = x + (size_t)idx * c;
  f16* d = out + (size_t)idx * GPAD;
  for (int i = 0; i < GPAD; ++i) d[i] = (i < c) ? (f16)s[i] : (f16)0.0f;
}

__global__ void concat_kernel(const float* __restrict__ feats,
                              const float* __restrict__ cents,
                              float* __restrict__ xn, int k, int co) {
  int idx = blockIdx.x * blockDim.x + threadIdx.x;
  if (idx >= NBATCH * k) return;
  const float* f = feats + (size_t)idx * co;
  const float* ce = cents + (size_t)idx * 3;
  float* dst = xn + (size_t)idx * (co + 3);
  for (int i = 0; i < co; ++i) dst[i] = f[i];
  dst[co] = ce[0]; dst[co + 1] = ce[1]; dst[co + 2] = ce[2];
}

// Final MLP: out = ((relu(h)@wf1+bf1) -> relu -> @wf2+bf2) @ wo + bo. 1 block/batch.
__global__ void head_mlp(const float* __restrict__ feats,
                         const float* __restrict__ wf1, const float* __restrict__ bf1,
                         const float* __restrict__ wf2, const float* __restrict__ bf2,
                         const float* __restrict__ wo, const float* __restrict__ bo,
                         float* __restrict__ out) {
  const int b = blockIdx.x, t = threadIdx.x;
  __shared__ float a1[256], h2[128], h3[64];
  for (int i = t; i < 256; i += 128) a1[i] = fmaxf(feats[b * 256 + i], 0.f);
  __syncthreads();
  if (t < 128) {
    float s = bf1[t];
    for (int kk = 0; kk < 256; ++kk) s += a1[kk] * wf1[kk * 128 + t];
    h2[t] = s;
  }
  __syncthreads();
  if (t < 64) {
    float s = bf2[t];
    for (int kk = 0; kk < 128; ++kk) s += fmaxf(h2[kk], 0.f) * wf2[kk * 64 + t];
    h3[t] = s;
  }
  __syncthreads();
  if (t == 0) {
    float s = bo[0];
    for (int kk = 0; kk < 64; ++kk) s += h3[kk] * wo[kk];
    out[b] = s;
  }
}

// ---------------------------------------------------------------- host
extern "C" void kernel_launch(void* const* d_in, const int* in_sizes, int n_in,
                              void* d_out, int out_size, void* d_ws, size_t ws_size,
                              hipStream_t stream) {
  (void)out_size; (void)ws_size;
  auto F = [&](int i) { return (const float*)d_in[i]; };

  // Input mapping: insertion-order dict flatten vs sorted-pytree flatten.
  const bool ins = (n_in >= 1 && in_sizes[0] == NBATCH * 512 * 3);
  const float* X = nullptr;
  const float *pw1[4], *pb1[4], *pwt1[4], *pbt1[4], *pwt2[4], *pbt2[4];
  const float *pw2[4], *pb2[4], *pw3[4], *pb3[4];
  const float *wf1, *bf1, *wf2, *bf2, *wo, *bo;
  if (ins) {
    X = F(0);
    for (int l = 0; l < 4; ++l) {
      int b = 1 + 10 * l;
      pw1[l] = F(b + 0); pb1[l] = F(b + 1); pwt1[l] = F(b + 2); pbt1[l] = F(b + 3);
      pwt2[l] = F(b + 4); pbt2[l] = F(b + 5); pw2[l] = F(b + 6); pb2[l] = F(b + 7);
      pw3[l] = F(b + 8); pb3[l] = F(b + 9);
    }
    wf1 = F(41); bf1 = F(42); wf2 = F(43); bf2 = F(44); wo = F(45); bo = F(46);
  } else {
    bf1 = F(0); bf2 = F(1); bo = F(2);
    for (int l = 0; l < 4; ++l) {
      int b = 3 + 10 * l;  // sorted keys: b1,b2,b3,bt1,bt2,w1,w2,w3,wt1,wt2
      pb1[l] = F(b + 0); pb2[l] = F(b + 1); pb3[l] = F(b + 2); pbt1[l] = F(b + 3);
      pbt2[l] = F(b + 4); pw1[l] = F(b + 5); pw2[l] = F(b + 6); pw3[l] = F(b + 7);
      pwt1[l] = F(b + 8); pwt2[l] = F(b + 9);
    }
    wf1 = F(43); wf2 = F(44); wo = F(45); X = F(46);
  }

  // Workspace carve
  char* ws = (char*)d_ws; size_t off = 0;
  auto alloc = [&](size_t bytes) -> void* {
    void* p = ws + off; off += (bytes + 255) & ~(size_t)255; return p;
  };
  float* x1 = (float*)alloc((size_t)NBATCH * 256 * 6 * 4);
  float* x2 = (float*)alloc((size_t)NBATCH * 128 * 9 * 4);
  float* x3 = (float*)alloc((size_t)NBATCH * 64 * 15 * 4);
  float* cents = (float*)alloc((size_t)NBATCH * 256 * 3 * 4);
  int* grp = (int*)alloc((size_t)NBATCH * 512 * 4);
  int* rnk = (int*)alloc((size_t)NBATCH * 512 * 4);
  int* cnts = (int*)alloc((size_t)NBATCH * 256 * 4);
  f16* grouped = (f16*)alloc((size_t)NBATCH * 256 * 512 * GPAD * 2);  // 8 MB max
  f16* w1t  = (f16*)alloc((size_t)64 * 32 * 2);
  f16* wt1t = (f16*)alloc((size_t)128 * 64 * 2);
  f16* wt2t = (f16*)alloc((size_t)4096 * 128 * 2);
  f16* w2t  = (f16*)alloc((size_t)128 * 64 * 2);
  f16* w3t  = (f16*)alloc((size_t)256 * 128 * 2);
  float* feats = (float*)alloc((size_t)NBATCH * 256 * 4);

  auto cdiv = [](int a, int b) { return (a + b - 1) / b; };
  const int ms[3] = {512, 256, 128};
  const int widths[3] = {3, 6, 9};
  const int couts[3] = {3, 6, 12};

  const float* xin = X;
  float* xnexts[3] = {x1, x2, x3};
  for (int l = 0; l < 3; ++l) {
    const int m = ms[l], k = m / 2, w = widths[l], co = couts[l];
    const int G = NBATCH * k;
    zero_u32<<<cdiv(NBATCH * k, 256), 256, 0, stream>>>((unsigned*)cnts, NBATCH * k);
    fps_kernel<<<NBATCH, 256, 0, stream>>>(xin, m, w, k, cents);
    assign_kernel<<<cdiv(NBATCH * m, 256), 256, 0, stream>>>(xin, cents, grp, rnk,
                                                             cnts, m, w, k);
    init_grouped<<<cdiv(G * m, 256), 256, 0, stream>>>(grouped, cents, k, m);
    scatter_kernel<<<cdiv(NBATCH * m, 256), 256, 0, stream>>>(xin, grp, rnk,
                                                              grouped, m, w, k);
    conv_w_t<<<cdiv(64 * 32, 256), 256, 0, stream>>>(pw1[l], w1t, w, 64, 32, 64);
    conv_w_t<<<cdiv(128 * 64, 256), 256, 0, stream>>>(pwt1[l], wt1t, 64, 128, 64, 128);
    conv_w_t<<<cdiv(4096 * 128, 256), 256, 0, stream>>>(pwt2[l], wt2t, 128, 4096, 128, 4096);
    conv_w_t<<<cdiv(128 * 64, 256), 256, 0, stream>>>(pw2[l], w2t, 64, 128, 64, 128);
    const int conp = ((co + 15) / 16) * 16;
    conv_w_t<<<cdiv(conp * 128, 256), 256, 0, stream>>>(pw3[l], w3t, 128, co, 128, conp);
    const size_t smem = (size_t)m * 64 * 2 + 64 * 64 * 2 + 4 * 16 * 128 * 2 +
                        128 * 4 + 128 * 4 + 256 * 4;
    if (co == 3)
      pointnet_kernel<3><<<G, 128, smem, stream>>>(grouped, pb1[l], pbt1[l],
          pbt2[l], pb2[l], pb3[l], w1t, wt1t, wt2t, w2t, w3t, feats, m);
    else if (co == 6)
      pointnet_kernel<6><<<G, 128, smem, stream>>>(grouped, pb1[l], pbt1[l],
          pbt2[l], pb2[l], pb3[l], w1t, wt1t, wt2t, w2t, w3t, feats, m);
    else
      pointnet_kernel<12><<<G, 128, smem, stream>>>(grouped, pb1[l], pbt1[l],
          pbt2[l], pb2[l], pb3[l], w1t, wt1t, wt2t, w2t, w3t, feats, m);
    concat_kernel<<<cdiv(G, 64), 64, 0, stream>>>(feats, cents, xnexts[l], k, co);
    xin = xnexts[l];
  }

  // Head PointNet: m=64, cin=15, cout=256, G=2 (no grouping)
  pack_rows_f16<<<cdiv(NBATCH * 64, 128), 128, 0, stream>>>(x3, grouped,
                                                            NBATCH * 64, 15);
  conv_w_t<<<cdiv(64 * 32, 256), 256, 0, stream>>>(pw1[3], w1t, 15, 64, 32, 64);
  conv_w_t<<<cdiv(128 * 64, 256), 256, 0, stream>>>(pwt1[3], wt1t, 64, 128, 64, 128);
  conv_w_t<<<cdiv(4096 * 128, 256), 256, 0, stream>>>(pwt2[3], wt2t, 128, 4096, 128, 4096);
  conv_w_t<<<cdiv(128 * 64, 256), 256, 0, stream>>>(pw2[3], w2t, 64, 128, 64, 128);
  conv_w_t<<<cdiv(256 * 128, 256), 256, 0, stream>>>(pw3[3], w3t, 128, 256, 128, 256);
  const size_t smemh = (size_t)64 * 64 * 2 + 64 * 64 * 2 + 4 * 16 * 128 * 2 +
                       128 * 4 + 128 * 4 + 256 * 4;
  pointnet_kernel<256><<<NBATCH, 128, smemh, stream>>>(grouped, pb1[3], pbt1[3],
      pbt2[3], pb2[3], pb3[3], w1t, wt1t, wt2t, w2t, w3t, feats, 64);
  head_mlp<<<NBATCH, 128, 0, stream>>>(feats, wf1, bf1, wf2, bf2, wo, bo,
                                       (float*)d_out);
}
